// MoE_52415780880436
// MI455X (gfx1250) — compile-verified, matrix operand order checked by reading, not run on hardware
//
#include <hip/hip_runtime.h>

// Problem constants (match the reference)
#define NTOK 8192          // B*T
#define DDIM 1024
#define NEXP 8
#define FDIM 1024
#define TM   32            // token-tile rows per block (2 WMMA M-subtiles)
#define MAXTILES (NTOK * 2 / TM + NEXP)   // 520, fixed grid for graph capture

typedef __attribute__((ext_vector_type(16))) __bf16 v16bf;
typedef __attribute__((ext_vector_type(8)))  __bf16 v8bf;
typedef __attribute__((ext_vector_type(8)))  float  v8f;

// -------------------------------------------------------------------------
// 0) zero per-expert counts (graph replays, so must re-zero every call)
__global__ void k_init(int* __restrict__ counts) {
  if (threadIdx.x < NEXP) counts[threadIdx.x] = 0;
}

// -------------------------------------------------------------------------
// 1) convert fp32 weights -> bf16, transposed so WMMA B operands are
//    K-contiguous in memory.
//    gating/up: in [e][d][f] -> out [e][f][d]
//    down:      in [e][f][d] -> out [e][d][f]
__global__ __launch_bounds__(256) void k_convert(
    const float* __restrict__ wg, const float* __restrict__ wu,
    const float* __restrict__ wd,
    __bf16* __restrict__ wg_t, __bf16* __restrict__ wu_t,
    __bf16* __restrict__ wd_t) {
  unsigned idx = blockIdx.x * 256u + threadIdx.x;   // < 3 * 2^23
  unsigned sel = idx >> 23;
  unsigned r   = idx & ((1u << 23) - 1u);
  unsigned e = r >> 20;
  unsigned a = (r >> 10) & 1023u;
  unsigned b = r & 1023u;
  unsigned in = (e << 20) + (b << 10) + a;
  const float* src = (sel == 0) ? wg : ((sel == 1) ? wu : wd);
  __bf16*      dst = (sel == 0) ? wg_t : ((sel == 1) ? wu_t : wd_t);
  dst[r] = (__bf16)src[in];
}

// -------------------------------------------------------------------------
// 2) router: one wave (32 lanes) per token.
__global__ __launch_bounds__(256) void k_router(
    const float* __restrict__ x, const float* __restrict__ wr,
    int* __restrict__ tE0, int* __restrict__ tE1,
    float* __restrict__ tW0, float* __restrict__ tW1,
    int* __restrict__ counts) {
  int wave = threadIdx.x >> 5;
  int lane = threadIdx.x & 31;
  int t = blockIdx.x * 8 + wave;
  float l[NEXP];
#pragma unroll
  for (int e = 0; e < NEXP; ++e) l[e] = 0.f;
  const float* xr = x + (size_t)t * DDIM;
  for (int d = lane; d < DDIM; d += 32) {
    float xv = xr[d];
#pragma unroll
    for (int e = 0; e < NEXP; ++e) l[e] += xv * wr[d * NEXP + e];
  }
#pragma unroll
  for (int off = 16; off > 0; off >>= 1) {
#pragma unroll
    for (int e = 0; e < NEXP; ++e) l[e] += __shfl_down(l[e], off, 32);
  }
  if (lane == 0) {
    float m = l[0];
#pragma unroll
    for (int e = 1; e < NEXP; ++e) m = fmaxf(m, l[e]);
    float p[NEXP], s = 0.f;
#pragma unroll
    for (int e = 0; e < NEXP; ++e) { p[e] = __expf(l[e] - m); s += p[e]; }
    float inv = 1.f / s;
#pragma unroll
    for (int e = 0; e < NEXP; ++e) p[e] *= inv;
    // top-2, ties -> lower index first (match jax.lax.top_k)
    int i0 = 0;
#pragma unroll
    for (int e = 1; e < NEXP; ++e) if (p[e] > p[i0]) i0 = e;
    int i1 = (i0 == 0) ? 1 : 0;
#pragma unroll
    for (int e = 0; e < NEXP; ++e) if (e != i0 && p[e] > p[i1]) i1 = e;
    // second softmax over the two probabilities (faithful to reference)
    float e1v = __expf(p[i1] - p[i0]);
    float w0 = 1.f / (1.f + e1v);
    float w1 = e1v / (1.f + e1v);
    tE0[t] = i0; tE1[t] = i1; tW0[t] = w0; tW1[t] = w1;
    atomicAdd(&counts[i0], 1);
    atomicAdd(&counts[i1], 1);
  }
}

// -------------------------------------------------------------------------
// 3) plan: offsets (padded to TM), tile map
__global__ void k_plan(const int* __restrict__ counts,
                       int* __restrict__ offsets, int* __restrict__ pcounts,
                       int* __restrict__ n_tiles,
                       int* __restrict__ tile_expert, int* __restrict__ tile_base) {
  if (threadIdx.x != 0 || blockIdx.x != 0) return;
  int off = 0, tile = 0;
  for (int e = 0; e < NEXP; ++e) {
    int c = counts[e];
    int nt = (c + TM - 1) / TM;
    offsets[e] = off;
    pcounts[e] = nt * TM;
    for (int i = 0; i < nt; ++i) {
      tile_expert[tile] = e;
      tile_base[tile]   = off + i * TM;
      ++tile;
    }
    off += nt * TM;
  }
  *n_tiles = tile;
}

// -------------------------------------------------------------------------
// 4) deterministic compaction: one block (1 wave) per expert, token order
__global__ void k_entries(const int* __restrict__ tE0, const int* __restrict__ tE1,
                          const float* __restrict__ tW0, const float* __restrict__ tW1,
                          const int* __restrict__ offsets, const int* __restrict__ pcounts,
                          int* __restrict__ entry_tok2, float* __restrict__ entry_w) {
  int e = blockIdx.x;
  int lane = threadIdx.x;
  int base = offsets[e];
  int cnt = 0;
  for (int t0 = 0; t0 < NTOK; t0 += 32) {
    int t = t0 + lane;
    int f0 = (tE0[t] == e) ? 1 : 0;
    int f1 = (tE1[t] == e) ? 1 : 0;
    int c = f0 + f1;
    int incl = c;
#pragma unroll
    for (int off = 1; off < 32; off <<= 1) {
      int v = __shfl_up(incl, off, 32);
      if (lane >= off) incl += v;
    }
    int excl = incl - c;
    if (f0) { entry_tok2[base + cnt + excl]      = t * 2 + 0; entry_w[base + cnt + excl]      = tW0[t]; }
    if (f1) { entry_tok2[base + cnt + excl + f0] = t * 2 + 1; entry_w[base + cnt + excl + f0] = tW1[t]; }
    cnt += __shfl(incl, 31, 32);
  }
  int pc = pcounts[e];
  for (int i = cnt + lane; i < pc; i += 32) {
    entry_tok2[base + i] = -1;
    entry_w[base + i] = 0.f;
  }
}

// -------------------------------------------------------------------------
// 5) expert MLP core: one block = TM(32)-token tile of one expert; 8 waves.
//    Each wave owns a 128-wide N chunk. Every B fragment (global bf16,
//    K-contiguous) feeds TWO WMMAs (two 16-row M-subtiles) to halve L2
//    weight traffic. Phase 1: gating/up GEMM (K=D) -> silu-fuse into LDS.
//    Phase 2: down GEMM (K=F), scale by router weight, write stage buffer.
__global__ __launch_bounds__(256) void k_mlp(
    const float* __restrict__ x,
    const __bf16* __restrict__ wg_t, const __bf16* __restrict__ wu_t,
    const __bf16* __restrict__ wd_t,
    const int* __restrict__ entry_tok2, const float* __restrict__ entry_w,
    const int* __restrict__ tile_expert, const int* __restrict__ tile_base,
    const int* __restrict__ n_tiles,
    float* __restrict__ stage) {
  __shared__ __bf16 sX[TM * DDIM];   // 64 KB: x tile (A of phase 1)
  __shared__ __bf16 sF[TM * DDIM];   // 64 KB: fused tile (A of phase 2)

  int tile = blockIdx.x;
  if (tile >= *n_tiles) return;
  int e    = tile_expert[tile];
  int base = tile_base[tile];

  // stage x rows (fp32 -> bf16, 16B vector writes); padded rows are zero
  for (int i = threadIdx.x; i < TM * DDIM / 8; i += 256) {
    int idx8 = i * 8;
    int m = idx8 >> 10;
    int d = idx8 & 1023;
    int tok2 = entry_tok2[base + m];
    v8bf v;
#pragma unroll
    for (int j = 0; j < 8; ++j) v[j] = (__bf16)0.f;
    if (tok2 >= 0) {
      const float* xp = x + (size_t)(tok2 >> 1) * DDIM + d;
#pragma unroll
      for (int j = 0; j < 8; ++j) v[j] = (__bf16)xp[j];
    }
    *(v8bf*)(sX + idx8) = v;
  }
  __syncthreads();

  int wave  = threadIdx.x >> 5;       // 0..7 -> 128-wide N chunk
  int lane  = threadIdx.x & 31;
  int lrow  = lane & 15;              // A row / B,C column within 16
  int lhalf = (lane >> 4) & 1;        // K-half selector per WMMA layout

  const __bf16* Bg = wg_t + (size_t)e * (FDIM * DDIM);
  const __bf16* Bu = wu_t + (size_t)e * (FDIM * DDIM);

  // ---- Phase 1: fuse = silu(x @ Wg) * (x @ Wu) ----
  for (int sub = 0; sub < 8; ++sub) {
    int n = wave * 128 + sub * 16 + lrow;     // global F column for this lane
    v8f accg0 = {}, accg1 = {};
    v8f accu0 = {}, accu1 = {};
    const __bf16* bgp = Bg + (size_t)n * DDIM;   // K-contiguous column data
    const __bf16* bup = Bu + (size_t)n * DDIM;
    const __bf16* ap0 = sX + lrow * DDIM;          // M-subtile 0 (rows 0..15)
    const __bf16* ap1 = sX + (16 + lrow) * DDIM;   // M-subtile 1 (rows 16..31)
#pragma unroll 4
    for (int k0 = 0; k0 < DDIM; k0 += 32) {
      int ka = k0 + lhalf * 8;
      v16bf a0, a1, bg, bu;
      *(v8bf*)&a0       = *(const v8bf*)(ap0 + ka);
      *((v8bf*)&a0 + 1) = *(const v8bf*)(ap0 + ka + 16);
      *(v8bf*)&a1       = *(const v8bf*)(ap1 + ka);
      *((v8bf*)&a1 + 1) = *(const v8bf*)(ap1 + ka + 16);
      *(v8bf*)&bg       = *(const v8bf*)(bgp + ka);
      *((v8bf*)&bg + 1) = *(const v8bf*)(bgp + ka + 16);
      *(v8bf*)&bu       = *(const v8bf*)(bup + ka);
      *((v8bf*)&bu + 1) = *(const v8bf*)(bup + ka + 16);
      accg0 = __builtin_amdgcn_wmma_f32_16x16x32_bf16(false, a0, false, bg,
                                                      (short)0, accg0, false, false);
      accg1 = __builtin_amdgcn_wmma_f32_16x16x32_bf16(false, a1, false, bg,
                                                      (short)0, accg1, false, false);
      accu0 = __builtin_amdgcn_wmma_f32_16x16x32_bf16(false, a0, false, bu,
                                                      (short)0, accu0, false, false);
      accu1 = __builtin_amdgcn_wmma_f32_16x16x32_bf16(false, a1, false, bu,
                                                      (short)0, accu1, false, false);
    }
#pragma unroll
    for (int r = 0; r < 8; ++r) {
      int m0 = lhalf * 8 + r;                 // C layout: row = r (+8 for hi lanes)
      float g0 = accg0[r], u0 = accu0[r];
      float g1 = accg1[r], u1 = accu1[r];
      sF[m0 * DDIM + n]        = (__bf16)((g0 / (1.f + __expf(-g0))) * u0);
      sF[(16 + m0) * DDIM + n] = (__bf16)((g1 / (1.f + __expf(-g1))) * u1);
    }
  }
  __syncthreads();

  // per-lane row metadata for the store (rows msub*16 + lhalf*8 + r)
  int   tok2s[16];
  float wsc[16];
#pragma unroll
  for (int msub = 0; msub < 2; ++msub) {
#pragma unroll
    for (int r = 0; r < 8; ++r) {
      int m = msub * 16 + lhalf * 8 + r;
      tok2s[msub * 8 + r] = entry_tok2[base + m];
      wsc[msub * 8 + r]   = entry_w[base + m];
    }
  }

  // ---- Phase 2: out = (fuse @ Wd) * router_weight, K = F ----
  const __bf16* Bd = wd_t + (size_t)e * (DDIM * FDIM);
  for (int sub = 0; sub < 8; ++sub) {
    int n = wave * 128 + sub * 16 + lrow;     // global D column
    v8f acc0 = {}, acc1 = {};
    const __bf16* bdp = Bd + (size_t)n * FDIM;
    const __bf16* ap0 = sF + lrow * FDIM;
    const __bf16* ap1 = sF + (16 + lrow) * FDIM;
#pragma unroll 4
    for (int k0 = 0; k0 < FDIM; k0 += 32) {
      int ka = k0 + lhalf * 8;
      v16bf a0, a1, b;
      *(v8bf*)&a0       = *(const v8bf*)(ap0 + ka);
      *((v8bf*)&a0 + 1) = *(const v8bf*)(ap0 + ka + 16);
      *(v8bf*)&a1       = *(const v8bf*)(ap1 + ka);
      *((v8bf*)&a1 + 1) = *(const v8bf*)(ap1 + ka + 16);
      *(v8bf*)&b        = *(const v8bf*)(bdp + ka);
      *((v8bf*)&b + 1)  = *(const v8bf*)(bdp + ka + 16);
      acc0 = __builtin_amdgcn_wmma_f32_16x16x32_bf16(false, a0, false, b,
                                                     (short)0, acc0, false, false);
      acc1 = __builtin_amdgcn_wmma_f32_16x16x32_bf16(false, a1, false, b,
                                                     (short)0, acc1, false, false);
    }
#pragma unroll
    for (int r = 0; r < 8; ++r) {
      int tok2 = tok2s[r];
      if (tok2 >= 0) stage[(size_t)tok2 * DDIM + n] = acc0[r] * wsc[r];
      int tok2b = tok2s[8 + r];
      if (tok2b >= 0) stage[(size_t)tok2b * DDIM + n] = acc1[r] * wsc[8 + r];
    }
  }
}

// -------------------------------------------------------------------------
// 6) combine the two expert-slot contributions per token (deterministic)
__global__ __launch_bounds__(256) void k_combine(const float* __restrict__ stage,
                                                 float* __restrict__ out) {
  int t = blockIdx.x;
  const float* s0 = stage + (size_t)(t * 2 + 0) * DDIM;
  const float* s1 = stage + (size_t)(t * 2 + 1) * DDIM;
  float* o = out + (size_t)t * DDIM;
  for (int d = threadIdx.x; d < DDIM; d += 256) o[d] = s0[d] + s1[d];
}

// -------------------------------------------------------------------------
extern "C" void kernel_launch(void* const* d_in, const int* in_sizes, int n_in,
                              void* d_out, int out_size, void* d_ws, size_t ws_size,
                              hipStream_t stream) {
  const float* x   = (const float*)d_in[0];
  const float* wr  = (const float*)d_in[1];
  const float* wgf = (const float*)d_in[2];
  const float* wuf = (const float*)d_in[3];
  const float* wdf = (const float*)d_in[4];
  float* out = (float*)d_out;
  char* ws = (char*)d_ws;

  // workspace layout (needs ~112.3 MB)
  __bf16* wg_t = (__bf16*)(ws + 0);
  __bf16* wu_t = (__bf16*)(ws + (16ull << 20));
  __bf16* wd_t = (__bf16*)(ws + (32ull << 20));
  float*  stage = (float*)(ws + (48ull << 20));          // [NTOK*2][DDIM] fp32
  char* meta = ws + (112ull << 20);
  int*   tE0 = (int*)(meta + 0);
  int*   tE1 = (int*)(meta + 32768);
  float* tW0 = (float*)(meta + 65536);
  float* tW1 = (float*)(meta + 98304);
  int* counts  = (int*)(meta + 131072);
  int* offsets = (int*)(meta + 131072 + 64);
  int* pcounts = (int*)(meta + 131072 + 128);
  int* n_tiles = (int*)(meta + 131072 + 192);
  int* tile_expert = (int*)(meta + 131072 + 256);
  int* tile_base   = (int*)(meta + 131072 + 256 + 4 * MAXTILES);
  int*   entry_tok2 = (int*)(meta + 131072 + 256 + 8 * MAXTILES);
  float* entry_w    = (float*)(meta + 131072 + 256 + 8 * MAXTILES + 4 * (NTOK * 2 + NEXP * TM));

  k_init<<<1, 32, 0, stream>>>(counts);
  k_convert<<<(3u << 23) / 256, 256, 0, stream>>>(wgf, wuf, wdf, wg_t, wu_t, wd_t);
  k_router<<<NTOK / 8, 256, 0, stream>>>(x, wr, tE0, tE1, tW0, tW1, counts);
  k_plan<<<1, 32, 0, stream>>>(counts, offsets, pcounts, n_tiles, tile_expert, tile_base);
  k_entries<<<NEXP, 32, 0, stream>>>(tE0, tE1, tW0, tW1, offsets, pcounts,
                                     entry_tok2, entry_w);
  k_mlp<<<MAXTILES, 256, 0, stream>>>(x, wg_t, wu_t, wd_t, entry_tok2, entry_w,
                                      tile_expert, tile_base, n_tiles, stage);
  k_combine<<<NTOK, 256, 0, stream>>>(stage, out);
}
